// FluidicBlock_36232344109586
// MI455X (gfx1250) — matrix operand order repaired and checked by reading, not compiled, not run on hardware
//
#include <hip/hip_runtime.h>

// FluidicBlock for MI455X (gfx1250): WMMA f16->f32 GEMMs, TDM-double-buffered
// persistent CfC scan.  B=4, S=4096, D=1024, N=128, H=512

typedef __attribute__((ext_vector_type(16))) _Float16 v16h;
typedef __attribute__((ext_vector_type(8)))  _Float16 v8h;
typedef __attribute__((ext_vector_type(8)))  float    v8f;

#define BB 4
#define SS 4096
#define DD 1024
#define NN 128
#define HH 512
#define KCAT (DD + HH) /* 1536 */
#define LN_EPSF 1e-5f
#define TDELTA 0.1f

__device__ __forceinline__ float sigmoidf_(float v) { return 1.0f / (1.0f + __expf(-v)); }

// --- Tensor Data Mover availability (device pass only) ----------------------
#if defined(__AMDGCN__) && __has_builtin(__builtin_amdgcn_tensor_load_to_lds) && \
    __has_builtin(__builtin_amdgcn_s_wait_tensorcnt)
#define HAVE_TDM 1
typedef unsigned int v4u __attribute__((ext_vector_type(4)));
typedef int          v4i __attribute__((ext_vector_type(4)));
typedef int          v8i __attribute__((ext_vector_type(8)));

// Async-load a 2D tile (BB rows x DD f16, row stride SS*DD elems) into LDS.
// D# bitfields per CDNA5 ISA ch.10.7/8: group0 {count=1, lds_addr, global_addr,
// type=2}, group1 {data_size=2B, tensor dims, tile dims, dim0 stride}.
__device__ __forceinline__ void tdm_load_xtile(const _Float16* gsrc, unsigned lds_addr) {
  unsigned long long ga = (unsigned long long)(size_t)gsrc;
  const unsigned long long stride0 = (unsigned long long)SS * DD; // data_size units
  v4u g0;
  g0.x = 1u;                                                  // count=1 (valid, user)
  g0.y = lds_addr;                                            // lds_addr[31:0]
  g0.z = (unsigned)(ga & 0xffffffffu);                        // global_addr[31:0]
  g0.w = (unsigned)((ga >> 32) & 0x01ffffffu) | (2u << 30);   // ga[56:32] | type=2
  v8i g1;
  g1[0] = 0x00010000;                                         // data_size=1 (2 bytes)
  g1[1] = (int)((DD & 0xffff) << 16);                         // tensor_dim0[15:0]
  g1[2] = (int)(((DD >> 16) & 0xffff) | ((BB & 0xffff) << 16)); // dim0 hi | tensor_dim1 lo
  g1[3] = (int)((DD & 0xffff) << 16);                         // tensor_dim1 hi | tile_dim0=DD
  g1[4] = BB;                                                 // tile_dim1=BB, tile_dim2=0
  g1[5] = (int)(stride0 & 0xffffffffu);                       // tensor_dim0_stride lo
  g1[6] = (int)((stride0 >> 32) & 0xffffu);                   // stride hi | dim1_stride lo
  g1[7] = 0;
  v4i g2 = {0, 0, 0, 0}, g3 = {0, 0, 0, 0};
#if __clang_major__ >= 23
  v8i g4 = {0, 0, 0, 0, 0, 0, 0, 0};
  __builtin_amdgcn_tensor_load_to_lds(g0, g1, g2, g3, g4, 0);
#else
  __builtin_amdgcn_tensor_load_to_lds(g0, g1, g2, g3, 0);
#endif
}
#endif

// ---------------- f32 -> f16 conversion (weights / staging) ----------------
__global__ void cvt_f32_to_f16(const float* __restrict__ src, _Float16* __restrict__ dst, int n) {
  int i = blockIdx.x * blockDim.x + threadIdx.x;
  if (i < n) dst[i] = (_Float16)src[i];
}

// ---------------- LayerNorm: one 256-thread block per row --------------------
__global__ void layernorm_kernel(const float* __restrict__ x,
                                 const float* __restrict__ g,
                                 const float* __restrict__ b,
                                 _Float16* __restrict__ xn) {
  __shared__ float s_sum[256], s_sq[256];
  int row = blockIdx.x;
  const float* xr = x + (size_t)row * DD;
  float sum = 0.f, sq = 0.f;
  for (int c = threadIdx.x; c < DD; c += 256) { float v = xr[c]; sum += v; sq += v * v; }
  s_sum[threadIdx.x] = sum; s_sq[threadIdx.x] = sq;
  __syncthreads();
  for (int off = 128; off > 0; off >>= 1) {
    if (threadIdx.x < off) {
      s_sum[threadIdx.x] += s_sum[threadIdx.x + off];
      s_sq[threadIdx.x]  += s_sq[threadIdx.x + off];
    }
    __syncthreads();
  }
  float mu   = s_sum[0] * (1.0f / DD);
  float var  = s_sq[0] * (1.0f / DD) - mu * mu;
  float rstd = rsqrtf(var + LN_EPSF);
  _Float16* xo = xn + (size_t)row * DD;
  for (int c = threadIdx.x; c < DD; c += 256)
    xo[c] = (_Float16)((xr[c] - mu) * rstd * g[c] + b[c]);
}

// ---------------- Generic WMMA GEMM: out = A(MxK,f16) * W(NxK,f16)^T ---------
__global__ void wmma_gemm_xwT(const _Float16* __restrict__ A,   // M x K
                              const _Float16* __restrict__ W,   // N x K
                              const float* __restrict__ bias,   // N (or null)
                              const float* __restrict__ resid,  // M x N f32 (or null)
                              float*    __restrict__ out32,     // M x N (or null)
                              _Float16* __restrict__ out16,     // M x N (or null)
                              int M, int N, int K) {
  int wid    = blockIdx.x * (blockDim.x >> 5) + (threadIdx.x >> 5);
  int ntiles = N >> 4;
  int tiles  = (M >> 4) * ntiles;
  if (wid >= tiles) return;                 // wave-uniform: EXEC all-1 where WMMA runs
  int mt = wid / ntiles, nt = wid % ntiles;
  int lane = threadIdx.x & 31;
  int half = lane >> 4, l16 = lane & 15;

  const _Float16* aRow = A + (size_t)(mt * 16 + l16) * K;
  const _Float16* wRow = W + (size_t)(nt * 16 + l16) * K + half * 16;

  v8f acc = {};
  for (int k = 0; k < K; k += 32) {
    if ((k & 255) == 0) {
      __builtin_prefetch(aRow + k + 256, 0, 3);   // WGP-scope global_prefetch_b8
      __builtin_prefetch(wRow + k + 256, 0, 3);
    }
    v8h a0 = *(const v8h*)(aRow + k + half * 16);
    v8h a1 = *(const v8h*)(aRow + k + half * 16 + 8);
    v16h av;
#pragma unroll
    for (int i = 0; i < 8; ++i) { av[i] = a0[i]; av[i + 8] = a1[i]; }
    v16h bv = *(const v16h*)(wRow + k);
    acc = __builtin_amdgcn_wmma_f32_16x16x32_f16(false, av, false, bv,
                                                 (short)0, acc, false, false);
  }
  int n = nt * 16 + l16;
  float bs = bias ? bias[n] : 0.0f;
#pragma unroll
  for (int r = 0; r < 8; ++r) {
    int m = mt * 16 + r + 8 * half;
    size_t idx = (size_t)m * N + n;
    float v = acc[r] + bs;
    if (resid) v += resid[idx];
    if (out32) out32[idx] = v;
    if (out16) out16[idx] = (_Float16)v;
  }
}

// ---------------- SSM scan: states[s] = A*states[s-1] + B_all[s] -------------
__global__ void ssm_scan_kernel(const float* __restrict__ Ball,     // B x S x N
                                const float* __restrict__ A_log,    // N
                                float* __restrict__ final_state,    // B x N (d_out)
                                _Float16* __restrict__ st16) {      // B x S x N
  int t = blockIdx.x * blockDim.x + threadIdx.x;
  if (t >= BB * NN) return;
  int b = t / NN, n = t % NN;
  float A = sigmoidf_(A_log[n]);
  float st = 0.0f;
  const float* src = Ball + (size_t)b * SS * NN + n;
  _Float16*    dst = st16 + (size_t)b * SS * NN + n;
  for (int s = 0; s < SS; ++s) {
    st = A * st + src[(size_t)s * NN];
    dst[(size_t)s * NN] = (_Float16)st;
  }
  final_state[t] = st;
}

// ---------------- Persistent CfC recurrence (single WGP, 32 waves) -----------
// x_t tile (4 x 1024 f16) is TDM-prefetched one step ahead into a double buffer.
// WMMA D-row m depends only on A-row m, so the 12 padding rows of the M=16 tile
// never need zeroing: lanes 4..15 alias batch row (l16 & 3).
__global__ void __launch_bounds__(1024)
cfc_scan_kernel(const _Float16* __restrict__ ssm16,  // B x S x D
                const _Float16* __restrict__ wcat,   // 4 x H x KCAT (ff1,ff2,ta,tb)
                const float* __restrict__ ff1_b, const float* __restrict__ ff2_b,
                const float* __restrict__ ta_b,  const float* __restrict__ tb_b,
                _Float16* __restrict__ cfc_seq,      // B x S x H
                float* __restrict__ h_last) {        // B x H (d_out)
  __shared__ __attribute__((aligned(32))) _Float16 xbuf[2][BB][DD]; // 16 KB, TDM dest
  __shared__ __attribute__((aligned(32))) _Float16 hbuf[BB][HH];    //  4 KB
  __shared__ __attribute__((aligned(32))) _Float16 res[4][BB][HH];  // 16 KB

  int tid = threadIdx.x;
  int wid = tid >> 5, lane = tid & 31, half = lane >> 4, l16 = lane & 15;
  int r4 = l16 & 3;   // batch row alias for A-fragment rows 4..15

  for (int i = tid; i < BB * HH; i += 1024) ((_Float16*)hbuf)[i] = (_Float16)0.0f;
  __syncthreads();

#ifdef HAVE_TDM
  if (wid == 0) tdm_load_xtile(ssm16, (unsigned)(size_t)&xbuf[0][0][0]); // prologue: x_0
#endif

  for (int s = 0; s < SS; ++s) {
    int buf = s & 1;
#ifdef HAVE_TDM
    if (wid == 0) __builtin_amdgcn_s_wait_tensorcnt(0);   // xbuf[buf] landed
    __syncthreads();
    if (wid == 0 && s + 1 < SS)                            // prefetch x_{s+1}
      tdm_load_xtile(ssm16 + (size_t)(s + 1) * DD, (unsigned)(size_t)&xbuf[buf ^ 1][0][0]);
#else
    for (int i = tid; i < BB * DD; i += 1024) {
      int m = i >> 10, c = i & (DD - 1);
      xbuf[buf][m][c] = ssm16[((size_t)m * SS + s) * DD + c];
    }
    __syncthreads();
#endif

#pragma unroll
    for (int j = 0; j < 4; ++j) {
      int t  = wid * 4 + j;
      int g  = t >> 5;        // which weight matrix (ff1,ff2,ta,tb)
      int nt = t & 31;        // n-tile within H=512
      const _Float16* wRow = wcat + ((size_t)g * HH + nt * 16 + l16) * KCAT + half * 16;
      const _Float16* xr = &xbuf[buf][r4][0];
      const _Float16* hr = &hbuf[r4][0];
      v8f acc = {};
      for (int k = 0; k < DD; k += 32) {                   // x part of cat
        v8h a0 = *(const v8h*)(xr + k + half * 16);
        v8h a1 = *(const v8h*)(xr + k + half * 16 + 8);
        v16h av;
#pragma unroll
        for (int i = 0; i < 8; ++i) { av[i] = a0[i]; av[i + 8] = a1[i]; }
        v16h bv = *(const v16h*)(wRow + k);
        acc = __builtin_amdgcn_wmma_f32_16x16x32_f16(false, av, false, bv,
                                                     (short)0, acc, false, false);
      }
      for (int k = 0; k < HH; k += 32) {                   // h part of cat
        v8h a0 = *(const v8h*)(hr + k + half * 16);
        v8h a1 = *(const v8h*)(hr + k + half * 16 + 8);
        v16h av;
#pragma unroll
        for (int i = 0; i < 8; ++i) { av[i] = a0[i]; av[i + 8] = a1[i]; }
        v16h bv = *(const v16h*)(wRow + DD + k);
        acc = __builtin_amdgcn_wmma_f32_16x16x32_f16(false, av, false, bv,
                                                     (short)0, acc, false, false);
      }
      const float* bp = (g == 0) ? ff1_b : (g == 1) ? ff2_b : (g == 2) ? ta_b : tb_b;
      float bs = bp[nt * 16 + l16];
      if (half == 0) {        // only rows 0..3 (real batch) matter
#pragma unroll
        for (int r = 0; r < BB; ++r)
          res[g][r][nt * 16 + l16] = (_Float16)(acc[r] + bs);
      }
    }
    __syncthreads();

    // h_new = tanh(ff1)*(1-t) + t*tanh(ff2),  t = sigmoid(ta*dt + tb)
    if (tid < BB * HH) {
      int m = tid >> 9, n = tid & (HH - 1);
      float f1 = tanhf((float)res[0][m][n]);
      float f2 = tanhf((float)res[1][m][n]);
      float ti = sigmoidf_((float)res[2][m][n] * TDELTA + (float)res[3][m][n]);
      float h  = f1 * (1.0f - ti) + ti * f2;
      hbuf[m][n] = (_Float16)h;                              // feeds next step
      cfc_seq[((size_t)m * SS + s) * HH + n] = (_Float16)h;  // for output GEMM
      if (s == SS - 1) h_last[m * HH + n] = h;
    }
    __syncthreads();
  }
}

// ---------------------------------------------------------------------------
extern "C" void kernel_launch(void* const* d_in, const int* in_sizes, int n_in,
                              void* d_out, int out_size, void* d_ws, size_t ws_size,
                              hipStream_t stream) {
  const float* x      = (const float*)d_in[0];
  const float* A_log  = (const float*)d_in[1];
  const float* B_w    = (const float*)d_in[2];
  const float* B_b    = (const float*)d_in[3];
  const float* C_w    = (const float*)d_in[4];
  const float* C_b    = (const float*)d_in[5];
  const float* norm_g = (const float*)d_in[6];
  const float* norm_b = (const float*)d_in[7];
  const float* ff1_w  = (const float*)d_in[8];
  const float* ff1_b  = (const float*)d_in[9];
  const float* ff2_w  = (const float*)d_in[10];
  const float* ff2_b  = (const float*)d_in[11];
  const float* ta_w   = (const float*)d_in[12];
  const float* ta_b   = (const float*)d_in[13];
  const float* tb_w   = (const float*)d_in[14];
  const float* tb_b   = (const float*)d_in[15];
  const float* out_w  = (const float*)d_in[16];
  const float* out_b  = (const float*)d_in[17];
  (void)in_sizes; (void)n_in; (void)out_size; (void)ws_size;

  float* layer_out   = (float*)d_out;                       // B*S*D
  float* final_state = layer_out + (size_t)BB * SS * DD;    // B*N
  float* h_last      = final_state + BB * NN;               // B*H

  char* ws = (char*)d_ws;
  size_t off = 0;
  auto alloc = [&](size_t bytes) -> char* {
    char* p = ws + off; off += (bytes + 255) & ~(size_t)255; return p;
  };
  _Float16* xn16   = (_Float16*)alloc((size_t)BB * SS * DD * 2);  // normalized x
  _Float16* Bw16   = (_Float16*)alloc((size_t)NN * DD * 2);
  float*    Ball   = (float*)   alloc((size_t)BB * SS * NN * 4);
  _Float16* st16   = (_Float16*)alloc((size_t)BB * SS * NN * 2);  // SSM states
  _Float16* Cw16   = (_Float16*)alloc((size_t)DD * NN * 2);
  _Float16* ssm16  = (_Float16*)alloc((size_t)BB * SS * DD * 2);  // ssm_out
  _Float16* wcat16 = (_Float16*)alloc((size_t)4 * HH * KCAT * 2); // ff1|ff2|ta|tb
  _Float16* ow16   = (_Float16*)alloc((size_t)DD * HH * 2);
  _Float16* cfc16  = (_Float16*)alloc((size_t)BB * SS * HH * 2);  // cfc sequence

  auto cvt = [&](const float* s, _Float16* d, int n) {
    cvt_f32_to_f16<<<(n + 255) / 256, 256, 0, stream>>>(s, d, n);
  };
  cvt(B_w,   Bw16, NN * DD);
  cvt(C_w,   Cw16, DD * NN);
  cvt(ff1_w, wcat16 + (size_t)0 * HH * KCAT, HH * KCAT);
  cvt(ff2_w, wcat16 + (size_t)1 * HH * KCAT, HH * KCAT);
  cvt(ta_w,  wcat16 + (size_t)2 * HH * KCAT, HH * KCAT);
  cvt(tb_w,  wcat16 + (size_t)3 * HH * KCAT, HH * KCAT);
  cvt(out_w, ow16, DD * HH);

  layernorm_kernel<<<BB * SS, 256, 0, stream>>>(x, norm_g, norm_b, xn16);

  // B_all = xn @ B_w^T + B_b   (M=16384, N=128, K=1024)
  {
    int tiles = (BB * SS / 16) * (NN / 16);
    wmma_gemm_xwT<<<tiles / 8, 256, 0, stream>>>(xn16, Bw16, B_b, nullptr,
                                                 Ball, nullptr, BB * SS, NN, DD);
  }

  ssm_scan_kernel<<<(BB * NN + 255) / 256, 256, 0, stream>>>(Ball, A_log,
                                                             final_state, st16);

  // ssm_out = states @ C_w^T + C_b  -> f16   (M=16384, N=1024, K=128)
  {
    int tiles = (BB * SS / 16) * (DD / 16);
    wmma_gemm_xwT<<<tiles / 8, 256, 0, stream>>>(st16, Cw16, C_b, nullptr,
                                                 nullptr, ssm16, BB * SS, DD, NN);
  }

  cfc_scan_kernel<<<1, 1024, 0, stream>>>(ssm16, wcat16, ff1_b, ff2_b, ta_b, tb_b,
                                          cfc16, h_last);

  // layer_out = x + cfc @ out_w^T + out_b   (M=16384, N=1024, K=512)
  {
    int tiles = (BB * SS / 16) * (DD / 16);
    wmma_gemm_xwT<<<tiles / 8, 256, 0, stream>>>(cfc16, ow16, out_b, x,
                                                 layer_out, nullptr, BB * SS, DD, HH);
  }
}
